// ASIFMultiHeadAttention_29506425323610
// MI455X (gfx1250) — matrix operand order repaired and checked by reading, not compiled, not run on hardware
//
#include <hip/hip_runtime.h>
#include <hip/hip_bf16.h>
#include <math.h>

#define B_  256
#define S_  200
#define D_  256
#define H_  4
#define HD_ 64
#define R_  16
#define BS_ (B_*S_)   // 51200
#define KP_ 224       // probs row stride: S padded to K-multiple of 32 (zeros)
#define SP_ 224       // value-tensor s-stride (pad never multiplied by nonzero)

typedef __attribute__((ext_vector_type(16))) _Float16 v16h;
typedef __attribute__((ext_vector_type(8)))  float    v8f;

union FragU { v16h h; uint4 q[2]; };

// ---------------------------------------------------------------------------
// Register-blocked WMMA macro-tile: one wave computes an (MT*16) x (NT*16)
// f32 tile of A(MxK,f16,row-major) @ B(KxN), B supplied transposed (Bt: NxK
// row-major) so every fragment load is a contiguous 16-byte global_load_b128.
// Per K-step: MT*2 + NT*2 unconditional loads feed MT*NT WMMAs.
//
// Edge handling WITHOUT conditional loads (avoids cndmask+flat_load codegen):
//  - rows/cols are ADDRESS-CLAMPED to M-1/N-1; a clamped (duplicate) row only
//    contributes to out-of-range output rows, which the store guards discard.
//  - K must be a multiple of 32 (caller pads with zeros where needed).
//
// Wave32 fragment layouts per CDNA5 ISA 7.12.2:
//   A 16x32 : lanes 0-15 -> K {k0..k0+7, k0+16..k0+23}; lanes 16-31 -> +8
//   B 32x16 : lanes 0-15 -> K {k0..k0+7, k0+16..k0+23}; lanes 16-31 -> +8
// ---------------------------------------------------------------------------
template<int MT, int NT>
__device__ __forceinline__ void wmma_block16(
    const _Float16* __restrict__ A, int lda,
    const _Float16* __restrict__ Bt, int ldb,
    int M, int N, int K, int row0, int col0,
    v8f (&acc)[MT][NT])
{
  const int lane = threadIdx.x & 31;
  const int hi   = lane >> 4;
  const int r    = lane & 15;
  const _Float16* Ap[MT];
  const _Float16* Bp[NT];
#pragma unroll
  for (int i = 0; i < MT; ++i) {
    int arow = row0 + i * 16 + r;
    arow = arow < M ? arow : M - 1;           // address clamp (memory safety)
    Ap[i] = A + (size_t)arow * lda;
  }
#pragma unroll
  for (int j = 0; j < NT; ++j) {
    int bcol = col0 + j * 16 + r;
    bcol = bcol < N ? bcol : N - 1;           // address clamp (memory safety)
    Bp[j] = Bt + (size_t)bcol * ldb;
  }
  for (int k0 = 0; k0 < K; k0 += 32) {
    FragU a[MT], b[NT];
    const int ka0 = k0 + hi * 8,  ka1 = ka0 + 16;  // A elems 0..7 / 8..15
    const int kb0 = k0 + hi * 8,  kb1 = kb0 + 16;  // B elems 0..7 / 8..15
#pragma unroll
    for (int i = 0; i < MT; ++i) {
      a[i].q[0] = *reinterpret_cast<const uint4*>(Ap[i] + ka0);
      a[i].q[1] = *reinterpret_cast<const uint4*>(Ap[i] + ka1);
    }
#pragma unroll
    for (int j = 0; j < NT; ++j) {
      b[j].q[0] = *reinterpret_cast<const uint4*>(Bp[j] + kb0);
      b[j].q[1] = *reinterpret_cast<const uint4*>(Bp[j] + kb1);
    }
#pragma unroll
    for (int i = 0; i < MT; ++i)
#pragma unroll
      for (int j = 0; j < NT; ++j)
        acc[i][j] = __builtin_amdgcn_wmma_f32_16x16x32_f16(
            false, a[i].h, false, b[j].h, (short)0, acc[i][j], false, false);
  }
}

// C/D layout: acc[j] <-> row = row0 + (lane>>4)*8 + j, col = col0 + (lane&15)

// ------------------------- conversion kernels ------------------------------
__global__ void k_cvt_f16(const float* __restrict__ in, _Float16* __restrict__ out,
                          size_t n) {
  size_t i = ((size_t)blockIdx.x * blockDim.x + threadIdx.x) * 4;
  if (i + 3 < n) {
    float4 v = *reinterpret_cast<const float4*>(in + i);
    out[i + 0] = (_Float16)v.x;
    out[i + 1] = (_Float16)v.y;
    out[i + 2] = (_Float16)v.z;
    out[i + 3] = (_Float16)v.w;
  }
}

// W is (K=D, N=D) row-major f32 ; store Wt f16 as (N, K) row-major
__global__ void k_cvt_wt(const float* __restrict__ W, _Float16* __restrict__ Wt) {
  int n = blockIdx.x, k = threadIdx.x;
  Wt[(size_t)n * D_ + k] = (_Float16)W[(size_t)k * D_ + n];
}

// ------------------------- projection GEMM ---------------------------------
// out = X(51200x256,f16) @ W(256x256) + bias.  vt==0: standard (B,S,D) f16.
// vt==1: value tensor stored transposed per head: (B,H,HD,SP_) f16 so the ctx
// GEMM's B-fragment loads are contiguous.
__global__ void k_gemm_proj(const _Float16* __restrict__ A,
                            const _Float16* __restrict__ Wt,
                            const float* __restrict__ bias,
                            _Float16* __restrict__ out, int vt) {
  constexpr int MT = 2, NT = 4;
  const int Mb = BS_ / (16 * MT);  // 1600
  const int Nb = D_ / (16 * NT);   // 4
  const int wave = blockIdx.x * (blockDim.x >> 5) + (threadIdx.x >> 5);
  if (wave >= Mb * Nb) return;
  const int mb = wave / Nb, nb = wave % Nb;
  const int row0 = mb * 16 * MT, col0 = nb * 16 * NT;
  v8f acc[MT][NT];
#pragma unroll
  for (int i = 0; i < MT; ++i)
#pragma unroll
    for (int j = 0; j < NT; ++j) acc[i][j] = {};
  wmma_block16<MT, NT>(A, D_, Wt, D_, BS_, D_, D_, row0, col0, acc);
  const int lane = threadIdx.x & 31, hi = lane >> 4, r = lane & 15;
#pragma unroll
  for (int ni = 0; ni < NT; ++ni) {
    const int col = col0 + ni * 16 + r;
    const float bvv = bias[col];
#pragma unroll
    for (int mi = 0; mi < MT; ++mi) {
#pragma unroll
      for (int j = 0; j < 8; ++j) {
        const int row = row0 + mi * 16 + hi * 8 + j;
        const float v = acc[mi][ni][j] + bvv;
        if (!vt) {
          out[(size_t)row * D_ + col] = (_Float16)v;
        } else {
          const int b = row / S_, s = row % S_;
          const int h = col >> 6, d = col & 63;
          out[(((size_t)b * H_ + h) * HD_ + d) * SP_ + s] = (_Float16)v;
        }
      }
    }
  }
}

// ------------------------- score GEMM --------------------------------------
// per (b,h): scores = q(200x64) @ k(200x64)^T -> f32 (200x200)
__global__ void k_gemm_scores(const _Float16* __restrict__ Qp,
                              const _Float16* __restrict__ Kp,
                              float* __restrict__ out) {
  constexpr int MT = 2, NT = 4;
  const int bh = blockIdx.z;
  const int b = bh / H_, h = bh % H_;
  const _Float16* A  = Qp + (size_t)b * S_ * D_ + (size_t)h * HD_;
  const _Float16* Bt = Kp + (size_t)b * S_ * D_ + (size_t)h * HD_;
  float* O = out + (size_t)bh * S_ * S_;
  const int Mb = (S_ + 16 * MT - 1) / (16 * MT);  // 7
  const int Nb = (S_ + 16 * NT - 1) / (16 * NT);  // 4
  const int wave = blockIdx.x * (blockDim.x >> 5) + (threadIdx.x >> 5);
  if (wave >= Mb * Nb) return;
  const int mb = wave / Nb, nb = wave % Nb;
  const int row0 = mb * 16 * MT, col0 = nb * 16 * NT;
  v8f acc[MT][NT];
#pragma unroll
  for (int i = 0; i < MT; ++i)
#pragma unroll
    for (int j = 0; j < NT; ++j) acc[i][j] = {};
  wmma_block16<MT, NT>(A, D_, Bt, D_, S_, S_, HD_, row0, col0, acc);
  const int lane = threadIdx.x & 31, hi = lane >> 4, r = lane & 15;
#pragma unroll
  for (int ni = 0; ni < NT; ++ni) {
    const int col = col0 + ni * 16 + r;
#pragma unroll
    for (int mi = 0; mi < MT; ++mi) {
#pragma unroll
      for (int j = 0; j < 8; ++j) {
        const int row = row0 + mi * 16 + hi * 8 + j;
        if (row < S_ && col < S_) O[(size_t)row * S_ + col] = acc[mi][ni][j];
      }
    }
  }
}

// ------------------------- softmax -----------------------------------------
// one wave32 per row; probs f16 with row stride KP_ and exact zeros in the
// K-pad columns [S_, KP_) so the ctx GEMM needs no K guard.
__global__ void k_softmax(const float* __restrict__ s1,
                          const float* __restrict__ s2,
                          const float* __restrict__ mask,
                          _Float16* __restrict__ probs) {
  const int row = blockIdx.x * 8 + (threadIdx.x >> 5);
  if (row >= B_ * H_ * S_) return;
  const int lane = threadIdx.x & 31;
  const int b = (row / S_) / H_;
  const float* p1 = s1 + (size_t)row * S_;
  const float* p2 = s2 ? s2 + (size_t)row * S_ : nullptr;
  const float* mk = mask + (size_t)b * S_;
  float v[7];
  float mx = -3.4e38f;
  for (int i = 0; i < 7; ++i) {
    const int k = lane + 32 * i;
    if (k < S_) {
      float s = p1[k];
      if (p2) s += p2[k];
      v[i] = s * 0.125f + mk[k];
      mx = fmaxf(mx, v[i]);
    } else v[i] = -3.4e38f;
  }
  for (int m = 16; m; m >>= 1) mx = fmaxf(mx, __shfl_xor(mx, m, 32));
  float sum = 0.f;
  for (int i = 0; i < 7; ++i) {
    const int k = lane + 32 * i;
    const float e = (k < S_) ? __expf(v[i] - mx) : 0.f;
    v[i] = e;
    sum += e;
  }
  for (int m = 16; m; m >>= 1) sum += __shfl_xor(sum, m, 32);
  const float inv = 1.f / sum;
  for (int i = 0; i < 7; ++i) {          // 7*32 == KP_, covers the pad too
    const int k = lane + 32 * i;
    probs[(size_t)row * KP_ + k] = (_Float16)((k < S_) ? v[i] * inv : 0.f);
  }
}

// ------------------------- context GEMM ------------------------------------
// per (b,h): ctx = probs(200xKP_,f16) @ V(200x64), V stored (HD,SP_) -> Bt.
// K = KP_ (multiple of 32, zero-padded probs). Output into ctx16 (B,S,D) at
// column offset h*64.
__global__ void k_gemm_ctx(const _Float16* __restrict__ probs,
                           const _Float16* __restrict__ Vt,
                           _Float16* __restrict__ ctx) {
  constexpr int MT = 2, NT = 4;
  const int bh = blockIdx.z;
  const int b = bh / H_, h = bh % H_;
  const _Float16* A  = probs + (size_t)bh * S_ * KP_;
  const _Float16* Bt = Vt + (size_t)bh * HD_ * SP_;
  _Float16* O = ctx + (size_t)b * S_ * D_ + (size_t)h * HD_;
  const int Mb = (S_ + 16 * MT - 1) / (16 * MT);  // 7
  const int Nb = HD_ / (16 * NT);                 // 1
  const int wave = blockIdx.x * (blockDim.x >> 5) + (threadIdx.x >> 5);
  if (wave >= Mb * Nb) return;
  const int mb = wave / Nb, nb = wave % Nb;
  const int row0 = mb * 16 * MT, col0 = nb * 16 * NT;
  v8f acc[MT][NT];
#pragma unroll
  for (int i = 0; i < MT; ++i)
#pragma unroll
    for (int j = 0; j < NT; ++j) acc[i][j] = {};
  wmma_block16<MT, NT>(A, KP_, Bt, SP_, S_, HD_, KP_, row0, col0, acc);
  const int lane = threadIdx.x & 31, hi = lane >> 4, r = lane & 15;
#pragma unroll
  for (int ni = 0; ni < NT; ++ni) {
    const int col = col0 + ni * 16 + r;
#pragma unroll
    for (int mi = 0; mi < MT; ++mi) {
#pragma unroll
      for (int j = 0; j < 8; ++j) {
        const int row = row0 + mi * 16 + hi * 8 + j;
        if (row < S_) O[(size_t)row * D_ + col] = (_Float16)acc[mi][ni][j];
      }
    }
  }
}

// ------------------------- output dense GEMM -------------------------------
// out_f32 = ctx16 @ W + bias + resid    (LayerNorm applied afterwards in-place)
__global__ void k_gemm_dense(const _Float16* __restrict__ A,
                             const _Float16* __restrict__ Wt,
                             const float* __restrict__ bias,
                             const float* __restrict__ resid,
                             float* __restrict__ out) {
  constexpr int MT = 2, NT = 4;
  const int Mb = BS_ / (16 * MT);  // 1600
  const int Nb = D_ / (16 * NT);   // 4
  const int wave = blockIdx.x * (blockDim.x >> 5) + (threadIdx.x >> 5);
  if (wave >= Mb * Nb) return;
  const int mb = wave / Nb, nb = wave % Nb;
  const int row0 = mb * 16 * MT, col0 = nb * 16 * NT;
  v8f acc[MT][NT];
#pragma unroll
  for (int i = 0; i < MT; ++i)
#pragma unroll
    for (int j = 0; j < NT; ++j) acc[i][j] = {};
  wmma_block16<MT, NT>(A, D_, Wt, D_, BS_, D_, D_, row0, col0, acc);
  const int lane = threadIdx.x & 31, hi = lane >> 4, r = lane & 15;
#pragma unroll
  for (int ni = 0; ni < NT; ++ni) {
    const int col = col0 + ni * 16 + r;
    const float bvv = bias[col];
#pragma unroll
    for (int mi = 0; mi < MT; ++mi) {
#pragma unroll
      for (int j = 0; j < 8; ++j) {
        const int row = row0 + mi * 16 + hi * 8 + j;
        out[(size_t)row * D_ + col] = acc[mi][ni][j] + bvv + resid[(size_t)row * D_ + col];
      }
    }
  }
}

// ------------------------- LayerNorm (in place, f32) -----------------------
__global__ void k_layernorm(float* __restrict__ y, const float* __restrict__ g,
                            const float* __restrict__ b) {
  const int row = blockIdx.x * 8 + (threadIdx.x >> 5);
  if (row >= BS_) return;
  const int lane = threadIdx.x & 31;
  float* p = y + (size_t)row * D_;
  float v[8];
  float s = 0.f;
  for (int i = 0; i < 8; ++i) { v[i] = p[lane + 32 * i]; s += v[i]; }
  for (int m = 16; m; m >>= 1) s += __shfl_xor(s, m, 32);
  const float mean = s * (1.f / D_);
  float var = 0.f;
  for (int i = 0; i < 8; ++i) { const float d = v[i] - mean; var += d * d; }
  for (int m = 16; m; m >>= 1) var += __shfl_xor(var, m, 32);
  const float inv = rsqrtf(var * (1.f / D_) + 1e-12f);
  for (int i = 0; i < 8; ++i) {
    const int k = lane + 32 * i;
    p[k] = (v[i] - mean) * inv * g[k] + b[k];
  }
}

// ------------------------- HIE ---------------------------------------------
// hr[b,r,d] = sum_n h[b,n,d] * Wr[n,r]
__global__ void k_hr(const float* __restrict__ h, const float* __restrict__ Wr,
                     float* __restrict__ hr) {
  const int b = blockIdx.x, r = blockIdx.y, d = threadIdx.x;
  const float* hb = h + (size_t)b * S_ * D_ + d;
  float acc = 0.f;
  for (int n = 0; n < S_; ++n) acc += hb[(size_t)n * D_] * Wr[n * R_ + r];
  hr[((size_t)b * R_ + r) * D_ + d] = acc;
}

__device__ __forceinline__ float qr_reduce(float v, float* sh) {
  const int t = threadIdx.x;
  sh[t] = v; __syncthreads();
  for (int off = 128; off > 0; off >>= 1) {
    if (t < off) sh[t] += sh[t + off];
    __syncthreads();
  }
  const float r = sh[0];
  __syncthreads();
  return r;
}

// per-batch modified Gram-Schmidt QR of hr^T (256x16) -> Q (B,D,R)
// (HIE output is invariant to column-sign choice, so MGS matches jnp QR)
__global__ void k_qr(const float* __restrict__ hr, float* __restrict__ Q) {
  const int b = blockIdx.x, d = threadIdx.x;
  __shared__ float sh[256];
  float c[R_];
  for (int j = 0; j < R_; ++j) c[j] = hr[((size_t)b * R_ + j) * D_ + d];
  for (int j = 0; j < R_; ++j) {
    for (int i = 0; i < j; ++i) {
      const float dot = qr_reduce(c[i] * c[j], sh);
      c[j] -= dot * c[i];
    }
    const float n2 = qr_reduce(c[j] * c[j], sh);
    c[j] *= rsqrtf(n2 + 1e-20f);
    Q[((size_t)b * D_ + d) * R_ + j] = c[j];
  }
}

// pX = h@Q, pA = hf@Q, pAt = (pX*pA>0)*pA ; one block per (b,n)
__global__ void k_pgate(const float* __restrict__ h, const float* __restrict__ hf,
                        const float* __restrict__ Q, float* __restrict__ pAt) {
  const int row = blockIdx.x;
  const int b = row / S_;
  const int t = threadIdx.x, r = t & 15, c = t >> 4;
  const float* hp  = h  + (size_t)row * D_;
  const float* hfp = hf + (size_t)row * D_;
  const float* Qb  = Q + (size_t)b * D_ * R_;
  float px = 0.f, pa = 0.f;
  for (int i = 0; i < 16; ++i) {
    const int d = c * 16 + i;
    const float q = Qb[(size_t)d * R_ + r];
    px += hp[d] * q;
    pa += hfp[d] * q;
  }
  __shared__ float sx[256], sa[256];
  sx[t] = px; sa[t] = pa; __syncthreads();
  if (c == 0) {
    for (int i = 1; i < 16; ++i) { px += sx[i * 16 + r]; pa += sa[i * 16 + r]; }
    pAt[(size_t)row * R_ + r] = (px * pa > 0.f) ? pa : 0.f;
  }
}

// h += pAt @ Q^T ; one block per (b,n), thread per d
__global__ void k_hie_add(float* __restrict__ h, const float* __restrict__ Q,
                          const float* __restrict__ pAt) {
  const int row = blockIdx.x;
  const int b = row / S_;
  const int d = threadIdx.x;
  __shared__ float sp[R_];
  if (threadIdx.x < R_) sp[threadIdx.x] = pAt[(size_t)row * R_ + threadIdx.x];
  __syncthreads();
  const float* Qr = Q + ((size_t)b * D_ + d) * R_;
  float acc = 0.f;
  for (int r = 0; r < R_; ++r) acc += sp[r] * Qr[r];
  h[(size_t)row * D_ + d] += acc;
}

// ===========================================================================
extern "C" void kernel_launch(void* const* d_in, const int* in_sizes, int n_in,
                              void* d_out, int out_size, void* d_ws, size_t ws_size,
                              hipStream_t stream) {
  const float* x_it = (const float*)d_in[0];
  const float* x_fu = (const float*)d_in[1];
  const float* x_po = (const float*)d_in[2];
  const float* mask = (const float*)d_in[3];
  const float* Wv  = (const float*)d_in[4];  const float* bv  = (const float*)d_in[5];
  const float* Wqf = (const float*)d_in[6];  const float* bqf = (const float*)d_in[7];
  const float* Wkf = (const float*)d_in[8];  const float* bkf = (const float*)d_in[9];
  const float* Wvf = (const float*)d_in[10]; const float* bvf = (const float*)d_in[11];
  const float* Wqp = (const float*)d_in[12]; const float* bqp = (const float*)d_in[13];
  const float* Wkp = (const float*)d_in[14]; const float* bkp = (const float*)d_in[15];
  const float* Wvp = (const float*)d_in[16]; const float* bvp = (const float*)d_in[17];
  const float* Wd  = (const float*)d_in[18]; const float* bd  = (const float*)d_in[19];
  const float* g1  = (const float*)d_in[20]; const float* b1  = (const float*)d_in[21];
  const float* Wfd = (const float*)d_in[22]; const float* bfd = (const float*)d_in[23];
  const float* g2  = (const float*)d_in[24]; const float* b2  = (const float*)d_in[25];
  const float* Wpd = (const float*)d_in[26]; const float* bpd = (const float*)d_in[27];
  const float* g3  = (const float*)d_in[28]; const float* b3  = (const float*)d_in[29];
  const float* Wr  = (const float*)d_in[30];

  float* out_h  = (float*)d_out;
  float* out_hf = out_h  + (size_t)BS_ * D_;
  float* out_hp = out_hf + (size_t)BS_ * D_;

  // ---- workspace carve-up ----
  char* w = (char*)d_ws;
  const size_t SZ_X16 = (size_t)BS_ * D_ * 2;            // 26.2 MB
  const size_t SZ_V16 = (size_t)B_ * H_ * HD_ * SP_ * 2; // 29.4 MB (padded V)
  const size_t SZ_W16 = (size_t)D_ * D_ * 2;
  const size_t SZ_SC  = (size_t)B_ * H_ * S_ * S_ * 4;   // 163.8 MB
  const size_t SZ_PR  = (size_t)B_ * H_ * S_ * KP_ * 2;  // 91.8 MB (padded K)
  auto take = [&](size_t sz) { char* p = w; w += (sz + 255) & ~(size_t)255; return p; };
  _Float16* x16_it = (_Float16*)take(SZ_X16);
  _Float16* x16_fu = (_Float16*)take(SZ_X16);
  _Float16* x16_po = (_Float16*)take(SZ_X16);
  _Float16* wt_v  = (_Float16*)take(SZ_W16);
  _Float16* wt_qf = (_Float16*)take(SZ_W16);
  _Float16* wt_kf = (_Float16*)take(SZ_W16);
  _Float16* wt_vf = (_Float16*)take(SZ_W16);
  _Float16* wt_qp = (_Float16*)take(SZ_W16);
  _Float16* wt_kp = (_Float16*)take(SZ_W16);
  _Float16* wt_vp = (_Float16*)take(SZ_W16);
  _Float16* wt_d  = (_Float16*)take(SZ_W16);
  _Float16* wt_fd = (_Float16*)take(SZ_W16);
  _Float16* wt_pd = (_Float16*)take(SZ_W16);
  _Float16* p_iv = (_Float16*)take(SZ_V16);  // item_v, (B,H,HD,SP_)
  _Float16* p_fq = (_Float16*)take(SZ_X16);
  _Float16* p_fk = (_Float16*)take(SZ_X16);
  _Float16* p_fv = (_Float16*)take(SZ_V16);  // (B,H,HD,SP_)
  _Float16* p_pq = (_Float16*)take(SZ_X16);
  _Float16* p_pk = (_Float16*)take(SZ_X16);
  _Float16* p_pv = (_Float16*)take(SZ_V16);  // (B,H,HD,SP_)
  float*    sc_item = (float*)take(SZ_SC);
  float*    sc_pos  = (float*)take(SZ_SC);
  _Float16* probs16 = (_Float16*)take(SZ_PR);
  _Float16* ctx16   = (_Float16*)take(SZ_X16);
  float*    hr_ws   = (float*)take((size_t)B_ * R_ * D_ * 4);
  float*    Q_ws    = (float*)take((size_t)B_ * D_ * R_ * 4);
  float*    pAt_ws  = (float*)take((size_t)BS_ * R_ * 4);

  // ---- 0) precision conversion ----
  {
    const size_t n = (size_t)BS_ * D_;
    const int blk = (int)(n / 4 / 256);
    k_cvt_f16<<<blk, 256, 0, stream>>>(x_it, x16_it, n);
    k_cvt_f16<<<blk, 256, 0, stream>>>(x_fu, x16_fu, n);
    k_cvt_f16<<<blk, 256, 0, stream>>>(x_po, x16_po, n);
  }
  k_cvt_wt<<<D_, D_, 0, stream>>>(Wv,  wt_v);
  k_cvt_wt<<<D_, D_, 0, stream>>>(Wqf, wt_qf);
  k_cvt_wt<<<D_, D_, 0, stream>>>(Wkf, wt_kf);
  k_cvt_wt<<<D_, D_, 0, stream>>>(Wvf, wt_vf);
  k_cvt_wt<<<D_, D_, 0, stream>>>(Wqp, wt_qp);
  k_cvt_wt<<<D_, D_, 0, stream>>>(Wkp, wt_kp);
  k_cvt_wt<<<D_, D_, 0, stream>>>(Wvp, wt_vp);
  k_cvt_wt<<<D_, D_, 0, stream>>>(Wd,  wt_d);
  k_cvt_wt<<<D_, D_, 0, stream>>>(Wfd, wt_fd);
  k_cvt_wt<<<D_, D_, 0, stream>>>(Wpd, wt_pd);

  // ---- 1) seven projection GEMMs (WMMA, 32x64 macro-tiles) ----
  const int PROJ_BLOCKS = (BS_ / 32) * (D_ / 64) / 4;  // 1600, 4 waves/block
  k_gemm_proj<<<PROJ_BLOCKS, 128, 0, stream>>>(x16_it, wt_v,  bv,  p_iv, 1);
  k_gemm_proj<<<PROJ_BLOCKS, 128, 0, stream>>>(x16_fu, wt_qf, bqf, p_fq, 0);
  k_gemm_proj<<<PROJ_BLOCKS, 128, 0, stream>>>(x16_fu, wt_kf, bkf, p_fk, 0);
  k_gemm_proj<<<PROJ_BLOCKS, 128, 0, stream>>>(x16_fu, wt_vf, bvf, p_fv, 1);
  k_gemm_proj<<<PROJ_BLOCKS, 128, 0, stream>>>(x16_po, wt_qp, bqp, p_pq, 0);
  k_gemm_proj<<<PROJ_BLOCKS, 128, 0, stream>>>(x16_po, wt_kp, bkp, p_pk, 0);
  k_gemm_proj<<<PROJ_BLOCKS, 128, 0, stream>>>(x16_po, wt_vp, bvp, p_pv, 1);

  // ---- 2) score GEMMs (WMMA), batched over (b,h) ----
  {
    const int Mb = (S_ + 31) / 32;  // 7
    const int Nb = (S_ + 63) / 64;  // 4
    dim3 g((Mb * Nb + 3) / 4, 1, B_ * H_);  // 7 blocks x 1024
    k_gemm_scores<<<g, 128, 0, stream>>>(p_fq, p_fk, sc_item);
    k_gemm_scores<<<g, 128, 0, stream>>>(p_pq, p_pk, sc_pos);
  }

  // ---- 3) three branches: softmax -> ctx (WMMA) -> dense (WMMA) -> LN ----
  const int SM_BLOCKS = (B_ * H_ * S_) / 8;       // 25600
  const int LN_BLOCKS = BS_ / 8;                  // 6400
  dim3 gctx((((S_ + 31) / 32) * (HD_ / 64) + 3) / 4, 1, B_ * H_);  // 2 x 1024

  // branch h: both_scores = item + pos
  k_softmax<<<SM_BLOCKS, 256, 0, stream>>>(sc_item, sc_pos, mask, probs16);
  k_gemm_ctx<<<gctx, 128, 0, stream>>>(probs16, p_iv, ctx16);
  k_gemm_dense<<<PROJ_BLOCKS, 128, 0, stream>>>(ctx16, wt_d, bd, x_it, out_h);
  k_layernorm<<<LN_BLOCKS, 256, 0, stream>>>(out_h, g1, b1);

  // branch hf: item scores only
  k_softmax<<<SM_BLOCKS, 256, 0, stream>>>(sc_item, nullptr, mask, probs16);
  k_gemm_ctx<<<gctx, 128, 0, stream>>>(probs16, p_fv, ctx16);
  k_gemm_dense<<<PROJ_BLOCKS, 128, 0, stream>>>(ctx16, wt_fd, bfd, x_fu, out_hf);
  k_layernorm<<<LN_BLOCKS, 256, 0, stream>>>(out_hf, g2, b2);

  // branch hp: pos scores only
  k_softmax<<<SM_BLOCKS, 256, 0, stream>>>(sc_pos, nullptr, mask, probs16);
  k_gemm_ctx<<<gctx, 128, 0, stream>>>(probs16, p_pv, ctx16);
  k_gemm_dense<<<PROJ_BLOCKS, 128, 0, stream>>>(ctx16, wt_pd, bpd, x_po, out_hp);
  k_layernorm<<<LN_BLOCKS, 256, 0, stream>>>(out_hp, g3, b3);

  // ---- 4) HIE: h += ((h@Q * hf@Q > 0) * hf@Q) @ Q^T, Q from QR(h^T Wr) ----
  k_hr<<<dim3(B_, R_), 256, 0, stream>>>(out_h, Wr, hr_ws);
  k_qr<<<B_, 256, 0, stream>>>(hr_ws, Q_ws);
  k_pgate<<<BS_, 256, 0, stream>>>(out_h, out_hf, Q_ws, pAt_ws);
  k_hie_add<<<BS_, 256, 0, stream>>>(out_h, Q_ws, pAt_ws);
}
